// MultiScaleHyperGenerator_13795434955276
// MI455X (gfx1250) — matrix operand order repaired and checked by reading, not compiled, not run on hardware
//
#include <hip/hip_runtime.h>
#include <hip/hip_bf16.h>
#include <math.h>

// ---------------- CDNA5 WMMA types ----------------
typedef __attribute__((ext_vector_type(16))) __bf16 v16bf;
typedef __attribute__((ext_vector_type(8)))  float  v8f;

union FragAB { v16bf v; uint4 q[2]; __bf16 e[16]; };

#define BATCH 16
#define NFC   64
#define NCIM  3
#define HIMG  256
#define PADW  7
#define HPAD  270   // 256 + 2*7

// norm_weight scales:
//  LR_GAIN = sqrt(2/(1+0.02^2)) = 1.41393087
#define SCALE_HEAD 0.27211113f    // LR_GAIN/sqrt(27)
#define SCALE_BODY 0.058913786f   // LR_GAIN/24
#define SCALE_TAIL 0.069444444f   // (5/3)/24
#define BSC_HEAD   0.57735027f    // 1/sqrt(3)
#define BSC_BODY   0.125f         // 1/sqrt(64)

__device__ __forceinline__ float lrelu(float x) { return fmaxf(x, 0.02f * x); }

// ------------------------------------------------------------------
// Async global->LDS tile copy (CDNA5 GLOBAL_LOAD_ASYNC_TO_LDS_B128,
// tracked with ASYNCcnt). Per-lane LDS dst offset = low 32 bits of the
// flat shared pointer (wave-relative LDS offset per aperture layout).
// ------------------------------------------------------------------
__device__ __forceinline__ void async_tile_load(const __bf16* __restrict__ src_row0,
                                                unsigned lds_base,
                                                int rowStrideElems,   // Win*64
                                                int chunksPerRow,     // (cols*64*2)/16
                                                int lane) {
  int total = 3 * chunksPerRow;
  for (int idx = lane; idx < total; idx += 32) {
    int row = idx / chunksPerRow, c16 = idx - row * chunksPerRow;
    const __bf16* gp = src_row0 + (size_t)row * rowStrideElems + (size_t)c16 * 8;
    unsigned lo = lds_base + (unsigned)(idx * 16);
    asm volatile("global_load_async_to_lds_b128 %0, %1, off"
                 :: "v"(lo), "v"(gp) : "memory");
  }
  asm volatile("s_wait_asynccnt 0x0" ::: "memory");
}

// ------------------------------------------------------------------
// Pad noise [16,3,256,256] (f32 planar) -> [16,270,270,4] bf16 channels-last
// ------------------------------------------------------------------
extern "C" __global__ void pad_kernel(const float* __restrict__ noise,
                                      __bf16* __restrict__ padbuf) {
  int t = blockIdx.x * blockDim.x + threadIdx.x;
  const int total = BATCH * HPAD * HPAD;
  if (t >= total) return;
  int c = t % HPAD;
  int r = (t / HPAD) % HPAD;
  int b = t / (HPAD * HPAD);
  bool interior = (r >= PADW) && (r < PADW + HIMG) && (c >= PADW) && (c < PADW + HIMG);
  union { __bf16 e[4]; uint2 u; } v;
  #pragma unroll
  for (int ch = 0; ch < 4; ++ch) {
    float x = 0.f;
    if (interior && ch < NCIM)
      x = noise[(((size_t)b * NCIM + ch) * HIMG + (r - PADW)) * HIMG + (c - PADW)];
    v.e[ch] = (__bf16)x;
  }
  *(uint2*)&padbuf[(size_t)t * 4] = v.u;
}

// ------------------------------------------------------------------
// Weight packing into exact per-lane WMMA B-fragment order.
// B fragment (32x16 KxN bf16): lane L -> col = L&15, h = L>>4;
// element j -> k_local = (j>>3)*16 + h*8 + (j&7).
// Channel interleave: WMMA tile nt, col c -> real channel n = c*4 + nt,
// so each lane's 4 accumulators are 4 consecutive channels (packed stores).
// ------------------------------------------------------------------
extern "C" __global__ void pack_head(const float* __restrict__ hw, __bf16* __restrict__ pw) {
  int t = blockIdx.x * blockDim.x + threadIdx.x;  // 16*4*32*16 = 32768 threads
  if (t >= 16 * 4 * 32 * 16) return;
  int j = t & 15, lane = (t >> 4) & 31, nt = (t >> 9) & 3, b = t >> 11;
  int h = lane >> 4, n = (lane & 15) * 4 + nt;
  int k = (j >> 3) * 16 + h * 8 + (j & 7);  // K order: k = tap*3 + ci, K=27 (pad->32)
  float val = 0.f;
  if (k < 27) {
    int tap = k / 3, ci = k - tap * 3;
    int dy = tap / 3, dx = tap - dy * 3;
    val = hw[((((size_t)b * NFC + n) * NCIM + ci) * 3 + dy) * 3 + dx] * SCALE_HEAD;
  }
  pw[t] = (__bf16)val;
}

extern "C" __global__ void pack_body(const float* __restrict__ bw, __bf16* __restrict__ pw) {
  int t = blockIdx.x * blockDim.x + threadIdx.x;  // 5*16*18*4*32*16 = 2949120 threads
  if (t >= 5 * 16 * 18 * 4 * 32 * 16) return;
  int j = t & 15, lane = (t >> 4) & 31, nt = (t >> 9) & 3;
  int rest = t >> 11;           // lb*18 + s
  int s = rest % 18, lb = rest / 18;
  int b = lb & 15, l = lb >> 4;
  int h = lane >> 4, n = (lane & 15) * 4 + nt;
  int k = s * 32 + (j >> 3) * 16 + h * 8 + (j & 7);  // K order: k = tap*64 + ci, K=576
  int tap = k >> 6, ci = k & 63;
  int dy = tap / 3, dx = tap - dy * 3;
  float val = bw[(((((size_t)l * 16 + b) * NFC + n) * NFC + ci) * 3 + dy) * 3 + dx] * SCALE_BODY;
  pw[t] = (__bf16)val;
}

extern "C" __global__ void pack_tail(const float* __restrict__ tw, __bf16* __restrict__ pw) {
  int t = blockIdx.x * blockDim.x + threadIdx.x;  // 16*18*32*16 = 147456 threads
  if (t >= 16 * 18 * 32 * 16) return;
  int j = t & 15, lane = (t >> 4) & 31;
  int rest = t >> 9;            // b*18 + s
  int s = rest % 18, b = rest / 18;
  int h = lane >> 4, n = lane & 15;   // only n<3 valid, rest zero-padded
  int k = s * 32 + (j >> 3) * 16 + h * 8 + (j & 7);
  int tap = k >> 6, ci = k & 63;
  int dy = tap / 3, dx = tap - dy * 3;
  float val = 0.f;
  if (n < NCIM)
    val = tw[((((size_t)b * NCIM + n) * NFC + ci) * 3 + dy) * 3 + dx] * SCALE_TAIL;
  pw[t] = (__bf16)val;
}

extern "C" __global__ void pack_bias(const float* __restrict__ hb, const float* __restrict__ bb,
                                     const float* __restrict__ tb, float* __restrict__ biasH,
                                     float* __restrict__ biasB, float* __restrict__ biasT) {
  int t = blockIdx.x * blockDim.x + threadIdx.x;
  if (t < 1024) {                       // head: [16][64]
    biasH[t] = hb[t] * BSC_HEAD;
  } else if (t < 1024 + 5120) {         // body: [5][16][64]
    int i = t - 1024;
    biasB[i] = bb[i] * BSC_BODY;
  } else if (t < 1024 + 5120 + 256) {   // tail padded: [16][16]
    int i = t - 6144;
    int n = i & 15, b = i >> 4;
    biasT[i] = (n < NCIM) ? tb[b * NCIM + n] * BSC_BODY : 0.f;
  }
}

// ------------------------------------------------------------------
// A-fragment (ISA 16-bit A layout): lane -> m = L&15, h = L>>4;
// element e[2v+p] -> k_local = (v>>2)*16 + h*8 + (v&3)*2 + p.
// Channels-last LDS tile -> two aligned ds_load_b128 per fragment.
// ------------------------------------------------------------------

// head conv: [16,270,270,4] bf16 -> [16,268,268,64] bf16, K=27 (1 k-step), lrelu
extern "C" __global__ __launch_bounds__(32)
void conv_head(const __bf16* __restrict__ in, __bf16* __restrict__ out,
               const __bf16* __restrict__ pw, const float* __restrict__ bias) {
  const int Wout = 268, Win = HPAD;
  int b = blockIdx.z, py = blockIdx.y;
  int px0 = blockIdx.x * 16; if (px0 > Wout - 16) px0 = Wout - 16;
  int lane = threadIdx.x;
  __shared__ alignas(16) __bf16 tile[3 * 18 * 4];
  for (int idx = lane; idx < 54; idx += 32) {
    int row = idx / 18, col = idx - row * 18;
    const __bf16* src = in + (((size_t)b * Win + (py + row)) * Win + (px0 + col)) * 4;
    *(uint2*)&tile[(row * 18 + col) * 4] = *(const uint2*)src;
  }
  __syncthreads();
  int m = lane & 15, h = lane >> 4;
  FragAB a;
  #pragma unroll
  for (int v = 0; v < 8; ++v) {
    int kl0 = ((v >> 2) * 16) + h * 8 + ((v & 3) * 2);
    #pragma unroll
    for (int p = 0; p < 2; ++p) {
      int k = kl0 + p;
      __bf16 val = (__bf16)0.f;
      if (k < 27) {
        int tap = k / 3, ci = k - tap * 3;
        int dy = tap / 3, dx = tap - dy * 3;
        val = tile[((dy * 18) + (m + dx)) * 4 + ci];
      }
      a.e[2 * v + p] = val;
    }
  }
  const float* bb4 = bias + b * 64 + m * 4;
  v8f acc[4];
  #pragma unroll
  for (int nt = 0; nt < 4; ++nt) {
    float bv = bb4[nt];
    v8f z = {bv, bv, bv, bv, bv, bv, bv, bv};
    acc[nt] = z;
    const __bf16* wp = pw + ((size_t)(b * 4 + nt) * 32 + lane) * 16;
    FragAB w;
    w.q[0] = *(const uint4*)(wp);
    w.q[1] = *(const uint4*)(wp + 8);
    acc[nt] = __builtin_amdgcn_wmma_f32_16x16x32_bf16(false, a.v, false, w.v, (short)0,
                                                      acc[nt], false, false);
  }
  size_t obase = ((size_t)b * Wout * Wout + (size_t)py * Wout + px0) * 64;
  #pragma unroll
  for (int i = 0; i < 8; ++i) {
    int mm = i + 8 * h;
    union { __bf16 e[4]; uint2 u; } o;
    #pragma unroll
    for (int nt = 0; nt < 4; ++nt) o.e[nt] = (__bf16)lrelu(acc[nt][i]);
    *(uint2*)&out[obase + (size_t)mm * 64 + m * 4] = o.u;
  }
}

// body conv: [16,Win,Win,64] -> [16,Wout,Wout,64], 32 px x 64 ch per wave,
// K=576 (18 k-steps x 8 WMMA), lrelu
extern "C" __global__ __launch_bounds__(32)
void conv_body(const __bf16* __restrict__ in, __bf16* __restrict__ out,
               const __bf16* __restrict__ pw, const float* __restrict__ bias,
               int Win, int Wout) {
  int b = blockIdx.z, py = blockIdx.y;
  int px0 = blockIdx.x * 32; if (px0 > Wout - 32) px0 = Wout - 32;
  int lane = threadIdx.x;
  __shared__ alignas(16) __bf16 tile[3 * 34 * 64];   // 13056 B input tile
  async_tile_load(in + (((size_t)b * Win + py) * Win + px0) * 64,
                  (unsigned)(size_t)&tile[0], Win * 64, 272, lane);
  __syncthreads();
  int m = lane & 15, h = lane >> 4;
  const float* bb4 = bias + b * 64 + m * 4;
  v8f accA[4], accB[4];
  #pragma unroll
  for (int nt = 0; nt < 4; ++nt) {
    float bv = bb4[nt];
    v8f z = {bv, bv, bv, bv, bv, bv, bv, bv};
    accA[nt] = z;
    accB[nt] = z;
  }
  const __bf16* pwb = pw + (size_t)b * (18 * 4 * 32 * 16);
  #pragma unroll 2
  for (int s = 0; s < 18; ++s) {
    int tap = s >> 1, ci0 = (s & 1) * 32;
    int dy = tap / 3, dx = tap - dy * 3;
    const __bf16* r0 = &tile[(dy * 34 + m + dx) * 64 + ci0 + h * 8];
    FragAB a0, a1;
    a0.q[0] = *(const uint4*)(r0);
    a0.q[1] = *(const uint4*)(r0 + 16);
    const __bf16* r1 = r0 + 16 * 64;
    a1.q[0] = *(const uint4*)(r1);
    a1.q[1] = *(const uint4*)(r1 + 16);
    const __bf16* wp = pwb + ((size_t)s * 4 * 32 + lane) * 16;
    #pragma unroll
    for (int nt = 0; nt < 4; ++nt) {
      FragAB w;
      w.q[0] = *(const uint4*)(wp);
      w.q[1] = *(const uint4*)(wp + 8);
      accA[nt] = __builtin_amdgcn_wmma_f32_16x16x32_bf16(false, a0.v, false, w.v, (short)0,
                                                         accA[nt], false, false);
      accB[nt] = __builtin_amdgcn_wmma_f32_16x16x32_bf16(false, a1.v, false, w.v, (short)0,
                                                         accB[nt], false, false);
      wp += 32 * 16;
    }
  }
  size_t obase = ((size_t)b * Wout * Wout + (size_t)py * Wout + px0) * 64;
  #pragma unroll
  for (int i = 0; i < 8; ++i) {
    int mm = i + 8 * h;
    union { __bf16 e[4]; uint2 u; } oA, oB;
    #pragma unroll
    for (int nt = 0; nt < 4; ++nt) {
      oA.e[nt] = (__bf16)lrelu(accA[nt][i]);
      oB.e[nt] = (__bf16)lrelu(accB[nt][i]);
    }
    *(uint2*)&out[obase + (size_t)mm * 64 + m * 4] = oA.u;
    *(uint2*)&out[obase + (size_t)(mm + 16) * 64 + m * 4] = oB.u;
  }
}

// tail conv: [16,258,258,64] -> d_out [16,3,256,256] f32 planar, tanh
extern "C" __global__ __launch_bounds__(32)
void conv_tail(const __bf16* __restrict__ in, float* __restrict__ out,
               const __bf16* __restrict__ pw, const float* __restrict__ bias) {
  const int Win = 258, Wout = 256;
  int b = blockIdx.z, py = blockIdx.y;
  int px0 = blockIdx.x * 16;
  int lane = threadIdx.x;
  __shared__ alignas(16) __bf16 tile[3 * 18 * 64];
  async_tile_load(in + (((size_t)b * Win + py) * Win + px0) * 64,
                  (unsigned)(size_t)&tile[0], Win * 64, 144, lane);
  __syncthreads();
  int m = lane & 15, h = lane >> 4;
  float bv = bias[b * 16 + m];
  v8f acc = {bv, bv, bv, bv, bv, bv, bv, bv};
  const __bf16* pwb = pw + (size_t)b * (18 * 32 * 16);
  #pragma unroll 2
  for (int s = 0; s < 18; ++s) {
    int tap = s >> 1, ci0 = (s & 1) * 32;
    int dy = tap / 3, dx = tap - dy * 3;
    const __bf16* abase = &tile[(dy * 18 + m + dx) * 64 + ci0 + h * 8];
    FragAB a;
    a.q[0] = *(const uint4*)(abase);
    a.q[1] = *(const uint4*)(abase + 16);
    const __bf16* wp = pwb + ((size_t)s * 32 + lane) * 16;
    FragAB w;
    w.q[0] = *(const uint4*)(wp);
    w.q[1] = *(const uint4*)(wp + 8);
    acc = __builtin_amdgcn_wmma_f32_16x16x32_bf16(false, a.v, false, w.v, (short)0,
                                                  acc, false, false);
  }
  if (m < NCIM) {   // divergence only after all WMMAs
    #pragma unroll
    for (int i = 0; i < 8; ++i) {
      int mm = i + 8 * h;
      out[(((size_t)b * NCIM + m) * Wout + py) * Wout + px0 + mm] = tanhf(acc[i]);
    }
  }
}

// ------------------------------------------------------------------
extern "C" void kernel_launch(void* const* d_in, const int* in_sizes, int n_in,
                              void* d_out, int out_size, void* d_ws, size_t ws_size,
                              hipStream_t stream) {
  (void)in_sizes; (void)n_in; (void)out_size; (void)ws_size;
  const float* noise  = (const float*)d_in[0];
  const float* head_w = (const float*)d_in[1];
  const float* head_b = (const float*)d_in[2];
  const float* body_w = (const float*)d_in[3];
  const float* body_b = (const float*)d_in[4];
  const float* tail_w = (const float*)d_in[5];
  const float* tail_b = (const float*)d_in[6];

  char* ws = (char*)d_ws;
  size_t off = 0;
  auto take = [&](size_t bytes) {
    char* p = ws + off;
    off = (off + bytes + 255) & ~(size_t)255;
    return p;
  };
  __bf16* padbuf = (__bf16*)take((size_t)BATCH * HPAD * HPAD * 4 * 2);
  __bf16* act0   = (__bf16*)take((size_t)BATCH * 268 * 268 * 64 * 2);
  __bf16* act1   = (__bf16*)take((size_t)BATCH * 268 * 268 * 64 * 2);
  __bf16* pwH    = (__bf16*)take((size_t)16 * 4 * 32 * 16 * 2);
  __bf16* pwB    = (__bf16*)take((size_t)5 * 16 * 18 * 4 * 32 * 16 * 2);
  __bf16* pwT    = (__bf16*)take((size_t)16 * 18 * 32 * 16 * 2);
  float*  biasH  = (float*)take(1024 * 4);
  float*  biasB  = (float*)take(5120 * 4);
  float*  biasT  = (float*)take(256 * 4);

  { int total = BATCH * HPAD * HPAD;
    pad_kernel<<<(total + 255) / 256, 256, 0, stream>>>(noise, padbuf); }
  pack_head<<<(16 * 4 * 32 * 16) / 256, 256, 0, stream>>>(head_w, pwH);
  pack_body<<<(5 * 16 * 18 * 4 * 32 * 16) / 256, 256, 0, stream>>>(body_w, pwB);
  pack_tail<<<(16 * 18 * 32 * 16) / 256, 256, 0, stream>>>(tail_w, pwT);
  pack_bias<<<(6400 + 255) / 256, 256, 0, stream>>>(head_b, body_b, tail_b,
                                                    biasH, biasB, biasT);

  dim3 blk(32, 1, 1);
  conv_head<<<dim3(17, 268, 16), blk, 0, stream>>>(padbuf, act0, pwH, biasH);

  int Win = 268;
  __bf16* cur = act0; __bf16* nxt = act1;
  for (int l = 0; l < 5; ++l) {
    int Wout = Win - 2;
    conv_body<<<dim3((Wout + 31) / 32, Wout, 16), blk, 0, stream>>>(
        cur, nxt, pwB + (size_t)l * 16 * 18 * 4 * 32 * 16, biasB + l * 1024, Win, Wout);
    __bf16* t = cur; cur = nxt; nxt = t;
    Win = Wout;
  }
  conv_tail<<<dim3(16, 256, 16), blk, 0, stream>>>(cur, (float*)d_out, pwT, biasT);
}
